// RNNPolicy_59030030516788
// MI455X (gfx1250) — matrix operand order repaired and checked by reading, not compiled, not run on hardware
//
#include <hip/hip_runtime.h>
#include <hip/hip_bf16.h>

typedef __attribute__((ext_vector_type(16))) __bf16        v16bf;
typedef __attribute__((ext_vector_type(8)))  float         v8f;
typedef __attribute__((ext_vector_type(8)))  unsigned int  v8u;

#define T_STEPS 512
#define NNEUR   1024
#define OBS     64
#define ACT     32

// Packed B-fragment dword counts (each fragment: 32 lanes x 8 dwords = 1KB).
// W_in fragments are placed directly after W_rec so the main K-loop indexes
// a single flat fragment array: frag(kt, nt) at ws + (kt*64 + nt)*256, kt=0..33.
// W_out follows W_in, which also makes the branch-free kt+2==34 prefetch
// land in-bounds (last touched dword = 573,439 of 573,440).
#define NW_DW (32 * 64 * 256)   // W_rec^T : kt32 = 0..31, nt = 0..63
#define NI_DW ( 2 * 64 * 256)   // W_in^T  : kt32 = 32..33, nt = 0..63
#define NO_DW (32 *  2 * 256)   // W_out^T : kt32 = 0..31, nt = 0..1

__device__ __forceinline__ unsigned short f2bf(float f) {
  unsigned int u = __float_as_uint(f);
  u += 0x7FFFu + ((u >> 16) & 1u);            // round-to-nearest-even
  return (unsigned short)(u >> 16);
}
__device__ __forceinline__ unsigned int pack2bf(float a, float b) {
  return (unsigned int)f2bf(a) | ((unsigned int)f2bf(b) << 16);
}
// tanh(x) = 1 - 2/(exp(2x)+1); exp(2x) = exp2(x * 2*log2(e)).
// Saturates exactly to +-1 (exp2 -> inf/0), ~5 VALU ops, 2 of them TRANS.
__device__ __forceinline__ float fast_tanh(float x) {
  float e = __builtin_amdgcn_exp2f(x * 2.885390081777927f);
  return 1.0f - 2.0f * __builtin_amdgcn_rcpf(e + 1.0f);
}

// Load 8 consecutive 32x16 B fragments (fragrow..fragrow+7, one v8u per lane each) into registers.
__device__ __forceinline__ void loadB(v8u* dst, const unsigned int* wsl, int fragrow) {
  const unsigned int* p = wsl + (size_t)fragrow * 256;
#pragma unroll
  for (int j = 0; j < 8; ++j) dst[j] = *(const v8u*)(p + j * 256);
}

// ---------------------------------------------------------------------------
// Prep: pack W_rec^T, W_in^T, W_out^T (fp32 row-major) into bf16 B-fragments.
// B-fragment layout (16-bit, 32x16 K x N, wave32):
//   lanes 0-15 : N = lane,    K = 0..7, 16..23  (VGPR i holds K = 16*(i>>2)+2*(i&3), +1)
//   lanes 16-31: N = lane-16, K = 8..15, 24..31
// ---------------------------------------------------------------------------
__global__ __launch_bounds__(256) void pack_weights(
    const float* __restrict__ W_rec, const float* __restrict__ W_in,
    const float* __restrict__ W_out, unsigned int* __restrict__ ws) {
  int tid = blockIdx.x * 256 + threadIdx.x;
  if (tid >= NW_DW + NI_DW + NO_DW) return;

  int region, local;
  if (tid < NW_DW)              { region = 0; local = tid; }
  else if (tid < NW_DW + NI_DW) { region = 1; local = tid - NW_DW; }
  else                          { region = 2; local = tid - NW_DW - NI_DW; }

  int f    = local >> 8;                    // fragment index within region
  int r    = local & 255;
  int lane = r >> 3;
  int i    = r & 7;                         // VGPR index within fragment
  int kloc = ((lane >> 4) << 3) + ((i >> 2) << 4) + 2 * (i & 3);    // K offset within 32-wide tile
  int ncol = lane & 15;

  float a, b;
  if (region == 0) {              // W_rec [N=1024][K=1024], B[k][n] = W_rec[n][k]
    int kt32 = f >> 6, nt = f & 63;
    int n = nt * 16 + ncol;
    int k = kt32 * 32 + kloc;
    a = W_rec[n * NNEUR + k];
    b = W_rec[n * NNEUR + k + 1];
  } else if (region == 1) {       // W_in [N=1024][D=64], B[d][n] = W_in[n][d]
    int kt32 = f >> 6, nt = f & 63;
    int n = nt * 16 + ncol;
    int k = kt32 * 32 + kloc;
    a = W_in[n * OBS + k];
    b = W_in[n * OBS + k + 1];
  } else {                        // W_out [A=32][K=1024], B[k][a] = W_out[a][k]
    int kt32 = f >> 1, nt = f & 1;
    int ai = nt * 16 + ncol;
    int k  = kt32 * 32 + kloc;
    a = W_out[ai * NNEUR + k];
    b = W_out[ai * NNEUR + k + 1];
  }
  ws[tid] = pack2bf(a, b);
}

// ---------------------------------------------------------------------------
// Persistent RNN kernel. grid = 16 (one WG per 16-row batch tile), block = 256.
// frag[0..31]  : h in A-fragment layout (bf16), resident in LDS all 512 steps
// frag[32..33] : x_t in A-fragment layout, rebuilt each step
// A-fragment layout (16-bit, 16x32 M x K):
//   element (m, koff): lane = m + 16*((koff>>3)&1),
//                      vgpr = 4*(koff>>4)+((koff&7)>>1), half = koff&1.
// __launch_bounds__(256, 2): exactly 8 waves on 4 SIMDs -> 2 waves/SIMD is
// the true residency requirement; frees the VGPR budget for deep pipelining.
// ---------------------------------------------------------------------------
__global__ __launch_bounds__(256, 2) void rnn_scan(
    const float* __restrict__ inputs,      // [256][512][64]
    const float* __restrict__ h0,          // [1024]
    const unsigned int* __restrict__ ws,   // packed weights
    float* __restrict__ out) {             // [256][512][32]
  __shared__ v8u   frag[34 * 32];              // 34KB: h (32 tiles) + x (2 tiles)
  __shared__ float out_part[8 * 2 * 32 * 8];   // 16KB: per-wave output partials

  const int tid  = threadIdx.x;
  const int wave = tid >> 5;
  const int lane = tid & 31;
  const int b0   = blockIdx.x * 16;

  unsigned short* hp = (unsigned short*)frag;

  // init h from h0 (broadcast over batch rows), in A-fragment order
  for (int idx = tid; idx < 16 * NNEUR; idx += 256) {
    int k = idx & (NNEUR - 1), m = idx >> 10;
    int kt = k >> 5, koff = k & 31;
    int alane = m + (((koff >> 3) & 1) << 4);
    int v     = ((koff >> 4) << 2) + ((koff & 7) >> 1);
    hp[kt * 512 + alane * 16 + v * 2 + (koff & 1)] = f2bf(h0[k]);
  }
  __syncthreads();

  const v8f vzero = {0.f, 0.f, 0.f, 0.f, 0.f, 0.f, 0.f, 0.f};

  // software-pipelined x load: xv holds x(t) while x(t+1) streams in
  const int   mX  = tid >> 4;
  const int   k0X = (tid & 15) << 2;
  const float* xbase = inputs + ((size_t)(b0 + mX) * T_STEPS) * OBS + k0X;
  float4 xv = *(const float4*)xbase;           // t = 0

  // precomputed x-fragment store slot for this thread
  unsigned int* xp;
  {
    int kt = k0X >> 5, koff = k0X & 31;
    int alane = mX + (((koff >> 3) & 1) << 4);
    int v     = ((koff >> 4) << 2) + ((koff & 7) >> 1);
    xp = (unsigned int*)(frag + 32 * 32) + kt * 256 + alane * 8 + v;
  }

  const unsigned int* wsl = ws + lane * 8;     // lane-resolved fragment base
  const int nt0 = wave * 8;

  for (int t = 0; t < T_STEPS; ++t) {
    // ---- P1: scatter x(t) into bf16 A-fragments; issue load of x(t+1) ----
    xp[0] = pack2bf(xv.x, xv.y);
    xp[1] = pack2bf(xv.z, xv.w);
    float4 xv_next = xv;
    if (t + 1 < T_STEPS) xv_next = *(const float4*)(xbase + (size_t)(t + 1) * OBS);
    __syncthreads();                          // B1: x ready, h-writes visible

    // ---- P2: acc[M=16, N-tile j] over K = 1024(h) + 64(x) ----
    // Manual 2-deep software pipeline: while the 8 WMMAs of tile kt run,
    // all 16 global_load_b128 of tile kt+1 are already in flight.
    v8f acc[8];
#pragma unroll
    for (int j = 0; j < 8; ++j) acc[j] = vzero;

    v8u b0r[8], b1r[8];
    loadB(b0r, wsl, nt0);                     // B(kt=0)
    v8u a0 = frag[lane];                      // A(kt=0)
    v8u a1 = frag[32 + lane];                 // A(kt=1)

    for (int kt = 0; kt < 34; kt += 2) {      // 17 even iterations
      loadB(b1r, wsl, (kt + 1) * 64 + nt0);   // prefetch B(kt+1), kt+1 <= 33
      v16bf av0 = __builtin_bit_cast(v16bf, a0);
#pragma unroll
      for (int j = 0; j < 8; ++j)
        acc[j] = __builtin_amdgcn_wmma_f32_16x16x32_bf16(
            false, av0, false, __builtin_bit_cast(v16bf, b0r[j]),
            (short)0, acc[j], false, false);
      {
        int ka = (kt + 2 < 34) ? kt + 2 : 33;
        a0 = frag[ka * 32 + lane];            // prefetch A(kt+2)
      }
      loadB(b0r, wsl, (kt + 2) * 64 + nt0);   // prefetch B(kt+2); kt+2==34 on
                                              // the last iter: in-bounds, unused
      v16bf av1 = __builtin_bit_cast(v16bf, a1);
#pragma unroll
      for (int j = 0; j < 8; ++j)
        acc[j] = __builtin_amdgcn_wmma_f32_16x16x32_bf16(
            false, av1, false, __builtin_bit_cast(v16bf, b1r[j]),
            (short)0, acc[j], false, false);
      {
        int ka = (kt + 3 < 34) ? kt + 3 : 33;
        a1 = frag[ka * 32 + lane];            // prefetch A(kt+3)
      }
    }
    __syncthreads();                          // B2: everyone done reading h

    // ---- P3: tanh, scatter h_new into A-fragment layout in LDS ----
    {
      int n_local  = lane & 15;
      int lanehalf = (n_local >> 3) & 1;
      int halfsel  = n_local & 1;
      int mbase    = (lane >> 4) << 3;
#pragma unroll
      for (int j = 0; j < 8; ++j) {
        int nt   = nt0 + j;
        int kt32 = nt >> 1;
        int v_a  = ((nt & 1) << 2) + ((n_local & 7) >> 1);
#pragma unroll
        for (int v = 0; v < 8; ++v) {
          int alane = mbase + v + (lanehalf << 4);
          hp[kt32 * 512 + alane * 16 + v_a * 2 + halfsel] = f2bf(fast_tanh(acc[j][v]));
        }
      }
    }
    __syncthreads();                          // B3: h_new ready

    // ---- P4: out = h_new @ W_out^T, K split across waves (128 each) ----
    {
      // Opaque OFFSET (not pointer): blocks hoisting of the 8 W_out
      // fragments into persistent VGPRs while keeping `ws`'s provable
      // global address space (pointer-laundering forced FLAT loads, which
      // needlessly couple LOADcnt with DScnt and burn the LDS path).
      unsigned int woff = NW_DW + NI_DW;
      asm volatile("" : "+s"(woff));
      const unsigned int* wp = ws + woff;
      v8u bo[8];
#pragma unroll
      for (int q = 0; q < 4; ++q) {
        int kt = wave * 4 + q;
        bo[2 * q]     = *(const v8u*)(wp + (kt * 2 + 0) * 256 + lane * 8);
        bo[2 * q + 1] = *(const v8u*)(wp + (kt * 2 + 1) * 256 + lane * 8);
      }
      v8f o0 = vzero, o1 = vzero;
#pragma unroll
      for (int q = 0; q < 4; ++q) {
        int kt = wave * 4 + q;
        v16bf av = __builtin_bit_cast(v16bf, frag[kt * 32 + lane]);
        o0 = __builtin_amdgcn_wmma_f32_16x16x32_bf16(
            false, av, false, __builtin_bit_cast(v16bf, bo[2 * q]), (short)0, o0, false, false);
        o1 = __builtin_amdgcn_wmma_f32_16x16x32_bf16(
            false, av, false, __builtin_bit_cast(v16bf, bo[2 * q + 1]), (short)0, o1, false, false);
      }
      float* pp = &out_part[wave * 512 + lane * 8];
#pragma unroll
      for (int v = 0; v < 8; ++v) pp[v]       = o0[v];
#pragma unroll
      for (int v = 0; v < 8; ++v) pp[256 + v] = o1[v];
    }
    __syncthreads();                          // B4: partials stored

    // ---- P5: sum 8 per-wave partials, write out[b, t, :] ----
#pragma unroll
    for (int s = 0; s < 2; ++s) {
      int e  = tid * 2 + s;
      int nt = e >> 8;
      int l  = (e >> 3) & 31;
      int v  = e & 7;
      float sum = 0.f;
#pragma unroll
      for (int w = 0; w < 8; ++w) sum += out_part[w * 512 + nt * 256 + l * 8 + v];
      int m = ((l >> 4) << 3) + v;
      int a = (nt << 4) + (l & 15);
      out[(((size_t)(b0 + m)) * T_STEPS + t) * ACT + a] = sum;
    }
    xv = xv_next;
    // no trailing barrier needed: B1 of the next step orders P5 reads
    // against the next P4's writes, and P1 only touches the x tiles.
  }
}

// ---------------------------------------------------------------------------
extern "C" void kernel_launch(void* const* d_in, const int* in_sizes, int n_in,
                              void* d_out, int out_size, void* d_ws, size_t ws_size,
                              hipStream_t stream) {
  (void)in_sizes; (void)n_in; (void)out_size; (void)ws_size;
  const float* inputs = (const float*)d_in[0];   // [256,512,64]
  const float* W_rec  = (const float*)d_in[1];   // [1024,1024]
  const float* W_in   = (const float*)d_in[2];   // [1024,64]
  const float* W_out  = (const float*)d_in[3];   // [32,1024]
  const float* h0     = (const float*)d_in[4];   // [1024]
  unsigned int* ws    = (unsigned int*)d_ws;     // needs ~2.2 MB

  const int total_dw = NW_DW + NI_DW + NO_DW;
  pack_weights<<<(total_dw + 255) / 256, 256, 0, stream>>>(W_rec, W_in, W_out, ws);
  rnn_scan<<<16, 256, 0, stream>>>(inputs, h0, ws, (float*)d_out);
}
